// SceneFun3DPN2_13503377178961
// MI455X (gfx1250) — matrix operand order repaired and checked by reading, not compiled, not run on hardware
//
#include <hip/hip_runtime.h>
#include <hip/hip_bf16.h>

typedef __attribute__((ext_vector_type(16))) _Float16 v16h;
typedef __attribute__((ext_vector_type(8)))  _Float16 v8h;
typedef __attribute__((ext_vector_type(8)))  float    v8f;

#define EPS_BN 1e-5f

// ---------------- small utils ----------------
__global__ void zero_f32(float* p, int n) {
  int i = blockIdx.x * blockDim.x + threadIdx.x;
  if (i < n) p[i] = 0.f;
}

__global__ void copy_f32(const float* __restrict__ s, float* __restrict__ d, int n) {
  int i = blockIdx.x * blockDim.x + threadIdx.x;
  if (i < n) d[i] = s[i];
}

// W[K][Nn] f32 -> Wt16[Npad][Kpad] f16 TRANSPOSED + zero padded.
// Transposing here makes the GEMM's B-tile K-contiguous per output column,
// so B fragments become two ds_load_b128 per wave instead of 16 ds_load_u16.
__global__ void convert_w(const float* __restrict__ W, _Float16* __restrict__ Wt,
                          int K, int Nn, int Kpad, int Npad) {
  int i = blockIdx.x * blockDim.x + threadIdx.x;
  int tot = Npad * Kpad;
  if (i >= tot) return;
  int n = i / Kpad, k = i % Kpad;
  Wt[i] = (k < K && n < Nn) ? (_Float16)W[k * Nn + n] : (_Float16)0.f;
}

// ---------------- farthest point sampling ----------------
// One workgroup (1024 threads = 32 waves) per batch. dist kept in registers
// (16 points per thread); argmax via wave32 shuffles + tiny LDS.
__global__ void fps_kernel(const float* __restrict__ xyz, int* __restrict__ out_idx,
                           int N, int S) {
  __shared__ float wval[32];
  __shared__ int   widx[32];
  __shared__ int   s_far;
  int b = blockIdx.x, t = threadIdx.x, T = blockDim.x;
  const float* base = xyz + (size_t)b * N * 3;
  float dloc[16];
#pragma unroll
  for (int j = 0; j < 16; j++) dloc[j] = 1e10f;
  if (t == 0) s_far = 0;
  __syncthreads();
  int lane = t & 31, w = t >> 5;
  for (int s = 0; s < S; s++) {
    int far = s_far;
    if (t == 0) out_idx[b * S + s] = far;
    float cx = base[far * 3 + 0], cy = base[far * 3 + 1], cz = base[far * 3 + 2];
    float bm = -1.f; int bi = 0;
    for (int j = 0; j < 16; j++) {
      int i = t + j * T;
      if (i < N) {
        float dx = base[i * 3 + 0] - cx, dy = base[i * 3 + 1] - cy, dz = base[i * 3 + 2] - cz;
        float d = dx * dx + dy * dy + dz * dz;
        float dd = fminf(dloc[j], d);
        dloc[j] = dd;
        if (dd > bm) { bm = dd; bi = i; }
      }
    }
    for (int off = 16; off; off >>= 1) {
      float ov = __shfl_down(bm, off);
      int   oi = __shfl_down(bi, off);
      if (ov > bm) { bm = ov; bi = oi; }
    }
    if (lane == 0) { wval[w] = bm; widx[w] = bi; }
    __syncthreads();
    if (w == 0) {
      float v = wval[lane]; int ix = widx[lane];
      for (int off = 16; off; off >>= 1) {
        float ov = __shfl_down(v, off);
        int   oi = __shfl_down(ix, off);
        if (ov > v) { v = ov; ix = oi; }
      }
      if (lane == 0) s_far = ix;
    }
    __syncthreads();
  }
}

__global__ void gather_xyz(const float* __restrict__ xyz, const int* __restrict__ idx,
                           float* __restrict__ out, int N, int S, int total) {
  int i = blockIdx.x * blockDim.x + threadIdx.x;
  if (i >= total) return;
  int bs = i / 3, c = i % 3;
  int b = bs / S;
  out[i] = xyz[((size_t)b * N + idx[bs]) * 3 + c];
}

// ---------------- ball query (wave32 ballot compaction) ----------------
__global__ void query_ball(const float* __restrict__ xyz, const float* __restrict__ cent,
                           int* __restrict__ out, int N, int S, int ns, float r2) {
  int gw = (blockIdx.x * blockDim.x + threadIdx.x) >> 5;
  int lane = threadIdx.x & 31;
  int b = gw / S, s = gw % S;
  const float* base = xyz + (size_t)b * N * 3;
  int q = b * S + s;
  float cx = cent[q * 3 + 0], cy = cent[q * 3 + 1], cz = cent[q * 3 + 2];
  int* dst = out + (size_t)q * ns;
  int cnt = 0, first = 0;
  bool haveFirst = false;
  for (int b0 = 0; b0 < N && cnt < ns; b0 += 32) {
    int i = b0 + lane;  // N is a multiple of 32
    float dx = base[i * 3 + 0] - cx, dy = base[i * 3 + 1] - cy, dz = base[i * 3 + 2] - cz;
    bool ok = (dx * dx + dy * dy + dz * dz) <= r2;
    unsigned m = (unsigned)__ballot(ok);
    int pos = cnt + __popc(m & ((1u << lane) - 1u));
    if (ok && pos < ns) dst[pos] = i;
    if (!haveFirst && m) {
      int sl = __ffs(m) - 1;
      first = __shfl(i, sl);
      haveFirst = true;
    }
    cnt += __popc(m);
  }
  if (cnt > ns) cnt = ns;
  for (int j = cnt + lane; j < ns; j += 32) dst[j] = first;
}

// ---------------- grouping -> padded f16 A matrices ----------------
__global__ void group_xyz_only(const float* __restrict__ xyz, const float* __restrict__ cent,
                               const int* __restrict__ idx, _Float16* __restrict__ A,
                               int N, int S, int ns, int Kpad, int total) {
  int r = blockIdx.x * blockDim.x + threadIdx.x;
  if (r >= total) return;
  int bs = r / ns, b = bs / S;
  int i = idx[r];
  _Float16* d = A + (size_t)r * Kpad;
  d[0] = (_Float16)(xyz[((size_t)b * N + i) * 3 + 0] - cent[bs * 3 + 0]);
  d[1] = (_Float16)(xyz[((size_t)b * N + i) * 3 + 1] - cent[bs * 3 + 1]);
  d[2] = (_Float16)(xyz[((size_t)b * N + i) * 3 + 2] - cent[bs * 3 + 2]);
  for (int c = 3; c < Kpad; c++) d[c] = (_Float16)0.f;
}

__global__ void group_feat(const float* __restrict__ pts, const float* __restrict__ srcxyz,
                           const float* __restrict__ cent, const int* __restrict__ idx,
                           _Float16* __restrict__ A, int Nsrc, int S, int ns, int C,
                           int Kpad, int total) {
  int r = blockIdx.x * blockDim.x + threadIdx.x;
  if (r >= total) return;
  int bs = r / ns, b = bs / S;
  int i = idx[r];
  const float* p = pts + ((size_t)b * Nsrc + i) * C;
  _Float16* d = A + (size_t)r * Kpad;
  for (int c = 0; c < C; c++) d[c] = (_Float16)p[c];
  d[C + 0] = (_Float16)(srcxyz[((size_t)b * Nsrc + i) * 3 + 0] - cent[bs * 3 + 0]);
  d[C + 1] = (_Float16)(srcxyz[((size_t)b * Nsrc + i) * 3 + 1] - cent[bs * 3 + 1]);
  d[C + 2] = (_Float16)(srcxyz[((size_t)b * Nsrc + i) * 3 + 2] - cent[bs * 3 + 2]);
  for (int c = C + 3; c < Kpad; c++) d[c] = (_Float16)0.f;
}

// ---------------- WMMA GEMM ----------------
// C[M,Npad] = A[M,Kpad] * Wt^T + bias. Wt is [Npad][Kpad] (pre-transposed).
// Block = 256 threads = 8 waves, block tile 64x64: waves 4(M) x 2(N), each
// wave owns a 16x32 tile = two v_wmma_f32_16x16x32_f16 sharing one A fragment.
// Fragment loads are ds_load_b128 pairs:
//  - A frag (ISA 16-bit A 16x32): lanes 0-15 hold K 0-7|16-23, lanes 16-31
//    hold K 8-15|24-31 -> two contiguous 8-half runs per lane.
//  - B frag: lane = column, same K split as A (lanes 0-15 K 0-7|16-23, lanes 16-31 K 8-15|24-31) ->
//    two contiguous 8-half runs per lane in the K-major LDS tile.
__global__ void __launch_bounds__(256)
gemm_wmma(const _Float16* __restrict__ A, const _Float16* __restrict__ Wt,
          const float* __restrict__ bias, float* __restrict__ C,
          int M, int Kpad, int Npad, int Nout) {
  __shared__ _Float16 As[64][40];
  __shared__ _Float16 BsT[64][40];
  int tid = threadIdx.x, lane = tid & 31, wv = tid >> 5;
  int wm = wv & 3, wn = wv >> 2;
  int m0 = blockIdx.x * 64, n0 = blockIdx.y * 64;
  int l = lane & 15;
  bool hi = lane >= 16;
  v8f acc0 = {}, acc1 = {};
  int rowT = tid >> 2, seg = tid & 3;
  const _Float16* aG = A + (size_t)(m0 + rowT) * Kpad + seg * 8;
  const _Float16* bG = Wt + (size_t)(n0 + rowT) * Kpad + seg * 8;
  int kb  = hi ? 8 : 0;    // A-frag K base within 8-half run
  int kb2 = hi ? 8 : 0;    // B-frag K base
  for (int k0 = 0; k0 < Kpad; k0 += 32) {
    *(uint4*)&As[rowT][seg * 8]  = *(const uint4*)(aG + k0);
    *(uint4*)&BsT[rowT][seg * 8] = *(const uint4*)(bG + k0);
    if (k0 + 32 < Kpad) __builtin_prefetch(aG + k0 + 32, 0, 3);
    __syncthreads();
    v8h a0 = *(const v8h*)&As[wm * 16 + l][kb];
    v8h a1 = *(const v8h*)&As[wm * 16 + l][kb + 16];
    v16h af = __builtin_shufflevector(a0, a1, 0, 1, 2, 3, 4, 5, 6, 7,
                                      8, 9, 10, 11, 12, 13, 14, 15);
    v8h b0 = *(const v8h*)&BsT[wn * 32 + l][kb2];
    v8h b1 = *(const v8h*)&BsT[wn * 32 + l][kb2 + 16];
    v16h bf0 = __builtin_shufflevector(b0, b1, 0, 1, 2, 3, 4, 5, 6, 7,
                                       8, 9, 10, 11, 12, 13, 14, 15);
    v8h c0 = *(const v8h*)&BsT[wn * 32 + 16 + l][kb2];
    v8h c1 = *(const v8h*)&BsT[wn * 32 + 16 + l][kb2 + 16];
    v16h bf1 = __builtin_shufflevector(c0, c1, 0, 1, 2, 3, 4, 5, 6, 7,
                                       8, 9, 10, 11, 12, 13, 14, 15);
    acc0 = __builtin_amdgcn_wmma_f32_16x16x32_f16(false, af, false, bf0,
                                                  (short)0, acc0, false, false);
    acc1 = __builtin_amdgcn_wmma_f32_16x16x32_f16(false, af, false, bf1,
                                                  (short)0, acc1, false, false);
    __syncthreads();
  }
  // C/D layout: VGPR r -> M = r (lanes 0-15) / 8+r (lanes 16-31), N = lane&15
  int colG0 = n0 + wn * 32 + l;
  int colG1 = colG0 + 16;
  float bv0 = (colG0 < Nout) ? bias[colG0] : 0.f;
  float bv1 = (colG1 < Nout) ? bias[colG1] : 0.f;
  int rbase = m0 + wm * 16 + (hi ? 8 : 0);
#pragma unroll
  for (int r = 0; r < 8; r++) {
    C[(size_t)(rbase + r) * Npad + colG0] = acc0[r] + bv0;
    C[(size_t)(rbase + r) * Npad + colG1] = acc1[r] + bv1;
  }
}

// ---------------- batchnorm (global stats) + relu ----------------
__global__ void bn_stats(const float* __restrict__ C, float* __restrict__ stats,
                         int M, int Npad, int Nout) {
  int tid = threadIdx.x;
  int c0 = tid, c1 = tid + 256;
  float s0 = 0.f, q0 = 0.f, s1 = 0.f, q1 = 0.f;
  for (int r = blockIdx.x; r < M; r += gridDim.x) {
    const float* row = C + (size_t)r * Npad;
    if (c0 < Nout) { float v = row[c0]; s0 += v; q0 += v * v; }
    if (c1 < Nout) { float v = row[c1]; s1 += v; q1 += v * v; }
  }
  if (c0 < Nout) { atomicAdd(&stats[c0], s0); atomicAdd(&stats[Nout + c0], q0); }
  if (c1 < Nout) { atomicAdd(&stats[c1], s1); atomicAdd(&stats[Nout + c1], q1); }
}

// normalize+relu in place (f32) and emit padded f16 copy as next GEMM input
__global__ void bn_relu(float* __restrict__ C, const float* __restrict__ stats,
                        const float* __restrict__ g, const float* __restrict__ be,
                        _Float16* __restrict__ A16, int M, int Npad, int Nout,
                        int KN, float invM) {
  size_t i = (size_t)blockIdx.x * blockDim.x + threadIdx.x;
  size_t tot = (size_t)M * KN;
  if (i >= tot) return;
  int r = (int)(i / KN), c = (int)(i % KN);
  if (c >= Nout) { A16[i] = (_Float16)0.f; return; }
  float mu = stats[c] * invM;
  float var = stats[Nout + c] * invM - mu * mu;
  float x = C[(size_t)r * Npad + c];
  float y = (x - mu) * rsqrtf(var + EPS_BN) * g[c] + be[c];
  y = fmaxf(y, 0.f);
  C[(size_t)r * Npad + c] = y;
  A16[i] = (_Float16)y;
}

__global__ void maxpool(const float* __restrict__ C, float* __restrict__ out,
                        int BS, int ns, int Npad, int Nout, int Cout, int coff,
                        int total) {
  int i = blockIdx.x * blockDim.x + threadIdx.x;
  if (i >= total) return;
  int bs = i / Nout, c = i % Nout;
  const float* p = C + (size_t)bs * ns * Npad + c;
  float m = -1e30f;
  for (int j = 0; j < ns; j++) m = fmaxf(m, p[(size_t)j * Npad]);
  out[(size_t)bs * Cout + coff + c] = m;
}

// ---------------- feature propagation (3-NN interp) ----------------
__global__ void knn3(const float* __restrict__ xyz1, const float* __restrict__ xyz2,
                     int* __restrict__ idx3, float* __restrict__ w3,
                     int N1, int N2, int total) {
  int i = blockIdx.x * blockDim.x + threadIdx.x;
  if (i >= total) return;
  int b = i / N1;
  float px = xyz1[(size_t)i * 3 + 0], py = xyz1[(size_t)i * 3 + 1], pz = xyz1[(size_t)i * 3 + 2];
  float d0 = 1e30f, d1 = 1e30f, d2 = 1e30f;
  int i0 = 0, i1 = 0, i2 = 0;
  const float* b2 = xyz2 + (size_t)b * N2 * 3;
  for (int j = 0; j < N2; j++) {
    float dx = b2[j * 3 + 0] - px, dy = b2[j * 3 + 1] - py, dz = b2[j * 3 + 2] - pz;
    float d = dx * dx + dy * dy + dz * dz;
    if (d < d0)      { d2 = d1; i2 = i1; d1 = d0; i1 = i0; d0 = d; i0 = j; }
    else if (d < d1) { d2 = d1; i2 = i1; d1 = d;  i1 = j; }
    else if (d < d2) { d2 = d;  i2 = j; }
  }
  float w0 = 1.f / (d0 + 1e-8f), w1 = 1.f / (d1 + 1e-8f), w2 = 1.f / (d2 + 1e-8f);
  float ws = w0 + w1 + w2;
  idx3[i * 3 + 0] = i0; idx3[i * 3 + 1] = i1; idx3[i * 3 + 2] = i2;
  w3[i * 3 + 0] = w0 / ws; w3[i * 3 + 1] = w1 / ws; w3[i * 3 + 2] = w2 / ws;
}

__global__ void fp_concat(const float* __restrict__ p1, const float* __restrict__ p2,
                          const int* __restrict__ idx3, const float* __restrict__ w3,
                          _Float16* __restrict__ A, int N1, int N2, int C1, int C2,
                          int Kpad, int total) {
  int r = blockIdx.x * blockDim.x + threadIdx.x;
  if (r >= total) return;
  int b = r / N1;
  _Float16* d = A + (size_t)r * Kpad;
  const float* s1 = p1 + (size_t)r * C1;
  for (int c = 0; c < C1; c++) d[c] = (_Float16)s1[c];
  const float* q0 = p2 + ((size_t)b * N2 + idx3[r * 3 + 0]) * C2;
  const float* q1 = p2 + ((size_t)b * N2 + idx3[r * 3 + 1]) * C2;
  const float* q2 = p2 + ((size_t)b * N2 + idx3[r * 3 + 2]) * C2;
  float w0 = w3[r * 3 + 0], w1 = w3[r * 3 + 1], w2 = w3[r * 3 + 2];
  for (int c = 0; c < C2; c++)
    d[C1 + c] = (_Float16)(w0 * q0[c] + w1 * q1[c] + w2 * q2[c]);
  for (int c = C1 + C2; c < Kpad; c++) d[c] = (_Float16)0.f;
}

// (B,N,20) -> (B,20,N)
__global__ void final_out(const float* __restrict__ C, float* __restrict__ out,
                          int N, int Npad, int Cout, int total) {
  int i = blockIdx.x * blockDim.x + threadIdx.x;
  if (i >= total) return;
  int b = i / (Cout * N);
  int rem = i % (Cout * N);
  int c = rem / N, n = rem % N;
  out[i] = C[((size_t)b * N + n) * Npad + c];
}

// ---------------- host orchestration ----------------
extern "C" void kernel_launch(void* const* d_in, const int* in_sizes, int n_in,
                              void* d_out, int out_size, void* d_ws, size_t ws_size,
                              hipStream_t stream) {
  const int B = 4, N = 16384, S1 = 512, S2 = 128;
  const float* xyz = (const float*)d_in[0];

  // Layers in setup_inputs() dict insertion order: sa1(b0,b1,b2), sa2, fp2, fp1, final.
  // Each layer dict: W, b, g, be.
  struct L { const float *W, *b, *g, *be; int K, Nn; };
  static const int chans[19][2] = {
    {3,32},{32,32},{32,64},
    {3,64},{64,64},{64,128},
    {3,64},{64,96},{96,128},
    {323,256},{256,256},{256,512},
    {832,256},{256,256},
    {259,256},{256,128},{128,128},
    {128,128},{128,20} };
  L ly[19];
  int p = 1;
  for (int i = 0; i < 19; i++) {
    ly[i].K = chans[i][0]; ly[i].Nn = chans[i][1];
    ly[i].W  = (const float*)d_in[p++];
    ly[i].b  = (const float*)d_in[p++];
    ly[i].g  = (const float*)d_in[p++];
    ly[i].be = (const float*)d_in[p++];
  }

  // workspace arena
  char* wsb = (char*)d_ws;
  size_t off = 0;
  auto alloc = [&](size_t bytes) -> void* {
    off = (off + 255) & ~(size_t)255;
    void* r = wsb + off;
    off += bytes;
    return r;
  };
  _Float16* A16 = (_Float16*)alloc((size_t)262144 * 96 * 2);   // max M*Kpad
  float*    C32 = (float*)alloc((size_t)262144 * 128 * 4);     // max M*Npad
  _Float16* W16 = (_Float16*)alloc((size_t)832 * 512 * 2);     // max Npad*Kpad
  float*  stats = (float*)alloc(1024 * 4);
  int*     fps1 = (int*)alloc((size_t)B * S1 * 4);
  int*     fps2 = (int*)alloc((size_t)B * S2 * 4);
  float*    l1x = (float*)alloc((size_t)B * S1 * 3 * 4);
  float*    l2x = (float*)alloc((size_t)B * S2 * 3 * 4);
  int*     ball = (int*)alloc((size_t)B * S1 * 128 * 4);
  float*    l1p = (float*)alloc((size_t)B * S1 * 320 * 4);
  float*    l2p = (float*)alloc((size_t)B * S2 * 512 * 4);
  float*   l1p2 = (float*)alloc((size_t)B * S1 * 256 * 4);
  int*     knnI = (int*)alloc((size_t)B * N * 3 * 4);
  float*   knnW = (float*)alloc((size_t)B * N * 3 * 4);

  auto round32 = [](int x) { return (x + 31) & ~31; };
  auto round64 = [](int x) { return (x + 63) & ~63; };
  auto runLayer = [&](const L& l, int M, bool bn) {
    int Kpad = round32(l.K), Npad = round64(l.Nn);
    int tot = Npad * Kpad;
    convert_w<<<(tot + 255) / 256, 256, 0, stream>>>(l.W, W16, l.K, l.Nn, Kpad, Npad);
    dim3 g(M / 64, Npad / 64);
    gemm_wmma<<<g, 256, 0, stream>>>(A16, W16, l.b, C32, M, Kpad, Npad, l.Nn);
    if (bn) {
      zero_f32<<<(2 * l.Nn + 255) / 256, 256, 0, stream>>>(stats, 2 * l.Nn);
      bn_stats<<<256, 256, 0, stream>>>(C32, stats, M, Npad, l.Nn);
      int KN = round32(l.Nn);
      size_t tot2 = (size_t)M * KN;
      bn_relu<<<(unsigned)((tot2 + 255) / 256), 256, 0, stream>>>(
          C32, stats, l.g, l.be, A16, M, Npad, l.Nn, KN, 1.f / (float)M);
    }
  };

  // ---- SA1 (multi-scale grouping) ----
  fps_kernel<<<B, 1024, 0, stream>>>(xyz, fps1, N, S1);
  { int t = B * S1 * 3;
    gather_xyz<<<(t + 255) / 256, 256, 0, stream>>>(xyz, fps1, l1x, N, S1, t); }
  const float radii1[3] = {0.1f, 0.2f, 0.4f};
  const int   ns1[3]    = {32, 64, 128};
  const int   loff[3]   = {0, 64, 192};   // out channels: 64 + 128 + 128 = 320
  for (int br = 0; br < 3; br++) {
    int ns = ns1[br];
    float r2 = radii1[br] * radii1[br];
    query_ball<<<(B * S1) / 8, 256, 0, stream>>>(xyz, l1x, ball, N, S1, ns, r2);
    int rows = B * S1 * ns;
    group_xyz_only<<<(rows + 255) / 256, 256, 0, stream>>>(xyz, l1x, ball, A16, N, S1, ns, 32, rows);
    for (int li = 0; li < 3; li++) runLayer(ly[br * 3 + li], rows, true);
    const L& last = ly[br * 3 + 2];
    int Npad = round64(last.Nn);
    int t = B * S1 * last.Nn;
    maxpool<<<(t + 255) / 256, 256, 0, stream>>>(C32, l1p, B * S1, ns, Npad, last.Nn, 320, loff[br], t);
  }

  // ---- SA2 ----
  fps_kernel<<<B, 1024, 0, stream>>>(l1x, fps2, S1, S2);
  { int t = B * S2 * 3;
    gather_xyz<<<(t + 255) / 256, 256, 0, stream>>>(l1x, fps2, l2x, S1, S2, t); }
  {
    int ns = 64; float r2 = 0.4f * 0.4f;
    query_ball<<<(B * S2) / 8, 256, 0, stream>>>(l1x, l2x, ball, S1, S2, ns, r2);
    int rows = B * S2 * ns;
    group_feat<<<(rows + 255) / 256, 256, 0, stream>>>(l1p, l1x, l2x, ball, A16, S1, S2, ns, 320, 352, rows);
    for (int li = 9; li < 12; li++) runLayer(ly[li], rows, true);
    int t = B * S2 * 512;
    maxpool<<<(t + 255) / 256, 256, 0, stream>>>(C32, l2p, B * S2, ns, 512, 512, 512, 0, t);
  }

  // ---- FP2: interp l2 -> l1 (832 -> 256 -> 256) ----
  {
    int M = B * S1;
    knn3<<<(M + 255) / 256, 256, 0, stream>>>(l1x, l2x, knnI, knnW, S1, S2, M);
    fp_concat<<<(M + 255) / 256, 256, 0, stream>>>(l1p, l2p, knnI, knnW, A16, S1, S2, 320, 512, 832, M);
    runLayer(ly[12], M, true);
    runLayer(ly[13], M, true);
    int t = M * 256;  // Npad == Nout == 256 -> contiguous
    copy_f32<<<(t + 255) / 256, 256, 0, stream>>>(C32, l1p2, t);
  }

  // ---- FP1: interp l1 -> l0 (259 -> 256 -> 128 -> 128) ----
  {
    int M = B * N;
    knn3<<<(M + 255) / 256, 256, 0, stream>>>(xyz, l1x, knnI, knnW, N, S1, M);
    fp_concat<<<(M + 255) / 256, 256, 0, stream>>>(xyz, l1p2, knnI, knnW, A16, N, S1, 3, 256, 288, M);
    runLayer(ly[14], M, true);
    runLayer(ly[15], M, true);
    runLayer(ly[16], M, true);
  }

  // ---- final head: 128 -> 128 (BN+relu) -> 20, then (B,N,20) -> (B,20,N) ----
  {
    int M = B * N;
    runLayer(ly[17], M, true);
    runLayer(ly[18], M, false);
    int t = B * 20 * N;
    final_out<<<(t + 255) / 256, 256, 0, stream>>>(C32, (float*)d_out, N, 64, 20, t);
  }
}
